// SelfAttentionBlock_52166672777673
// MI455X (gfx1250) — compile-verified, matrix operand order checked by reading, not run on hardware
//
#include <hip/hip_runtime.h>
#include <hip/hip_bf16.h>

// ---------------------------------------------------------------------------
// SelfAttentionBlock on gfx1250 (MI455X): bf16 WMMA GEMMs + f32 softmax/LN.
// B=4, L=2048, D=1024.  All GEMMs run v_wmma_f32_16x16x32_bf16.
// Block tile 128x256, 8 waves as 2x4, each wave owns a 64x64 patch
// (4x4 accumulator tiles -> 16 WMMAs per 32-wide K step).
// Tiles are staged with CDNA5 async copies (global_load_async_to_lds_b128,
// ASYNCcnt) into double-buffered LDS: one barrier per K step, k+1 copy
// overlaps k's WMMAs, no VGPR staging.
// ---------------------------------------------------------------------------

typedef __attribute__((ext_vector_type(16))) __bf16 v16bf;
typedef __attribute__((ext_vector_type(8)))  float  v8f;

#define LDSK 40  // 32 K-elems + 8 pad (bf16 units); row = 80B, 16B-aligned
#define ABUF (128 * LDSK)   // elems per A buffer
#define BBUF (256 * LDSK)   // elems per B buffer

__device__ __forceinline__ __bf16 f2bf(float f) {
  unsigned u = __builtin_bit_cast(unsigned, f);
  unsigned r = u + 0x7FFFu + ((u >> 16) & 1u);   // round-to-nearest-even
  unsigned short h = (unsigned short)(r >> 16);
  return __builtin_bit_cast(__bf16, h);
}

// Low 32 bits of a generic shared pointer == LDS byte address (ISA 10.2).
__device__ __forceinline__ unsigned lds_off(const void* p) {
  return (unsigned)(unsigned long long)p;
}

// CDNA5 async copy: 16B global -> LDS per lane, tracked with ASYNCcnt.
__device__ __forceinline__ void async_b128(unsigned lds, const void* g) {
  asm volatile("global_load_async_to_lds_b128 %0, %1, off"
               :: "v"(lds), "v"(g) : "memory");
}
__device__ __forceinline__ void wait_async0() {
  asm volatile("s_wait_asynccnt 0" ::: "memory");
}

struct Frag { union { v16bf v; unsigned u[8]; }; };

// C[128x256] += A[128xK] * Bt[256xK]^T   (Bt = B transposed, [N][K] row-major)
__device__ __forceinline__ void mm_mainloop(
    const __bf16* __restrict__ A, int lda,
    const __bf16* __restrict__ Bt, int ldb, int K,
    int row0, int col0, __bf16* As, __bf16* Bs, v8f acc[4][4])
{
  const int tid  = threadIdx.x;
  const int lane = tid & 31;
  const int wave = tid >> 5;
  const int wr   = wave >> 2;          // 2 row groups of 64
  const int wc   = wave & 3;           // 4 col groups of 64
  const int r16  = lane & 15;
  const int h8   = (lane >> 4) << 3;   // 0 or 8
  int ko[8];
#pragma unroll
  for (int i = 0; i < 8; ++i)          // ISA 16-bit A/B fragment K offsets
    ko[i] = ((i >> 2) << 4) + h8 + ((i & 3) << 1);

  const int aRow = tid >> 1;           // 0..127  (A: 2 threads per row)
  const int aSeg = (tid & 1) << 4;     // 0 or 16 bf16 elems (32B)

  const __bf16* gA = A  + (long)(row0 + aRow) * lda + aSeg;   // this thread's A slot
  const __bf16* gB = Bt + (long)(col0 + tid) * ldb;           // this thread's B row
  const unsigned sA = lds_off(As) + (unsigned)(aRow * LDSK + aSeg) * 2u;
  const unsigned sB = lds_off(Bs) + (unsigned)(tid * LDSK) * 2u;

  // prologue: tile 0 -> buffer 0
  async_b128(sA,      gA);
  async_b128(sA + 16, gA + 8);
  async_b128(sB,      gB);
  async_b128(sB + 16, gB + 8);
  async_b128(sB + 32, gB + 16);
  async_b128(sB + 48, gB + 24);

  for (int k0 = 0; k0 < K; k0 += 32) {
    const int cur = (k0 >> 5) & 1;
    wait_async0();        // this wave's copies into buf[cur] have landed
    __syncthreads();      // everyone's copies landed; prior reads of buf[cur^1] done

    if (k0 + 32 < K) {    // kick off tile k+1 into the other buffer
      const int nxt = cur ^ 1;
      const __bf16* gA2 = gA + k0 + 32;
      const __bf16* gB2 = gB + k0 + 32;
      const unsigned dA = sA + (unsigned)(nxt * ABUF) * 2u;
      const unsigned dB = sB + (unsigned)(nxt * BBUF) * 2u;
      async_b128(dA,      gA2);
      async_b128(dA + 16, gA2 + 8);
      async_b128(dB,      gB2);
      async_b128(dB + 16, gB2 + 8);
      async_b128(dB + 32, gB2 + 16);
      async_b128(dB + 48, gB2 + 24);
    }

    const __bf16* Abuf = As + cur * ABUF;
    const __bf16* Bbuf = Bs + cur * BBUF;
    Frag a[4], b[4];
#pragma unroll
    for (int t = 0; t < 4; ++t) {
      const __bf16* base = Abuf + (wr * 64 + t * 16 + r16) * LDSK;
#pragma unroll
      for (int i = 0; i < 8; ++i) a[t].u[i] = *(const unsigned*)(base + ko[i]);
    }
#pragma unroll
    for (int c = 0; c < 4; ++c) {
      const __bf16* base = Bbuf + (wc * 64 + c * 16 + r16) * LDSK;
#pragma unroll
      for (int i = 0; i < 8; ++i) b[c].u[i] = *(const unsigned*)(base + ko[i]);
    }
#pragma unroll
    for (int t = 0; t < 4; ++t)
#pragma unroll
      for (int c = 0; c < 4; ++c)
        acc[t][c] = __builtin_amdgcn_wmma_f32_16x16x32_bf16(
            false, a[t].v, false, b[c].v, (short)0, acc[t][c], false, false);
    // no trailing barrier: frag loads are consumed by WMMAs (dscnt-waited)
    // before this wave reaches the next iteration's barrier.
  }
}

// ------------------------- conversion kernels ------------------------------
__global__ __launch_bounds__(256) void k_cvt_bf16(const float* __restrict__ in,
                                                  __bf16* __restrict__ out, long n) {
  long i = (long)blockIdx.x * 256 + threadIdx.x;
  if (i < n) out[i] = f2bf(in[i]);
}

// Wt[n][k] = bf16(W[k][n]);  W is [K][N] row-major
__global__ __launch_bounds__(256) void k_cvt_transpose(const float* __restrict__ W,
                                                       __bf16* __restrict__ Wt,
                                                       int K, int N) {
  long i = (long)blockIdx.x * 256 + threadIdx.x;
  long total = (long)K * N;
  if (i < total) {
    int n = (int)(i / K), k = (int)(i % K);
    Wt[i] = f2bf(W[(long)k * N + n]);
  }
}

// ------------------------- GEMM 1: QKV -------------------------------------
__global__ __launch_bounds__(256) void k_gemm_qkv(
    const __bf16* __restrict__ xb, const __bf16* __restrict__ wt,
    const float* __restrict__ bias,
    __bf16* __restrict__ qb, __bf16* __restrict__ kb, __bf16* __restrict__ vt)
{
  __shared__ __align__(16) __bf16 As[2 * ABUF];
  __shared__ __align__(16) __bf16 Bs[2 * BBUF];
  const int row0 = blockIdx.y * 128, col0 = blockIdx.x * 256;
  v8f acc[4][4] = {};
  mm_mainloop(xb, 1024, wt, 1024, 1024, row0, col0, As, Bs, acc);

  const int tid = threadIdx.x, lane = tid & 31, wave = tid >> 5;
  const int wr = wave >> 2, wc = wave & 3, r16 = lane & 15, h8 = (lane >> 4) << 3;
#pragma unroll
  for (int t = 0; t < 4; ++t)
#pragma unroll
    for (int c = 0; c < 4; ++c)
#pragma unroll
      for (int j = 0; j < 8; ++j) {
        const int row = row0 + wr * 64 + t * 16 + h8 + j;
        const int col = col0 + wc * 64 + c * 16 + r16;
        const __bf16 o = f2bf(acc[t][c][j] + bias[col]);
        if (col < 1024)      qb[(long)row * 1024 + col] = o;
        else if (col < 2048) kb[(long)row * 1024 + (col - 1024)] = o;
        else // V stored transposed: vt[b][d][l]
          vt[((long)(row >> 11) * 1024 + (col - 2048)) * 2048 + (row & 2047)] = o;
      }
}

// ------------------------- GEMM 2: S = Q K^T -------------------------------
__global__ __launch_bounds__(256) void k_gemm_scores(
    const __bf16* __restrict__ qb, const __bf16* __restrict__ kb,
    float* __restrict__ S)
{
  __shared__ __align__(16) __bf16 As[2 * ABUF];
  __shared__ __align__(16) __bf16 Bs[2 * BBUF];
  const long boffA = (long)blockIdx.z * 2048 * 1024;
  const long boffS = (long)blockIdx.z * 2048 * 2048;
  const int row0 = blockIdx.y * 128, col0 = blockIdx.x * 256;
  v8f acc[4][4] = {};
  mm_mainloop(qb + boffA, 1024, kb + boffA, 1024, 1024, row0, col0, As, Bs, acc);

  const int tid = threadIdx.x, lane = tid & 31, wave = tid >> 5;
  const int wr = wave >> 2, wc = wave & 3, r16 = lane & 15, h8 = (lane >> 4) << 3;
#pragma unroll
  for (int t = 0; t < 4; ++t)
#pragma unroll
    for (int c = 0; c < 4; ++c)
#pragma unroll
      for (int j = 0; j < 8; ++j) {
        const int row = row0 + wr * 64 + t * 16 + h8 + j;
        const int col = col0 + wc * 64 + c * 16 + r16;
        S[boffS + (long)row * 2048 + col] = acc[t][c][j];
      }
}

// ------------------------- softmax (scale folded in) -----------------------
__global__ __launch_bounds__(256) void k_softmax(const float* __restrict__ S,
                                                 __bf16* __restrict__ P, float scale)
{
  __shared__ float red[256];
  const int tid = threadIdx.x;
  const long base = (long)blockIdx.x * 2048;
  float v[8];
#pragma unroll
  for (int i = 0; i < 8; ++i) v[i] = S[base + tid + i * 256] * scale;
  float m = v[0];
#pragma unroll
  for (int i = 1; i < 8; ++i) m = fmaxf(m, v[i]);
  red[tid] = m; __syncthreads();
  for (int s = 128; s > 0; s >>= 1) {
    if (tid < s) red[tid] = fmaxf(red[tid], red[tid + s]);
    __syncthreads();
  }
  m = red[0]; __syncthreads();
  float sum = 0.f;
#pragma unroll
  for (int i = 0; i < 8; ++i) { v[i] = __expf(v[i] - m); sum += v[i]; }
  red[tid] = sum; __syncthreads();
  for (int s = 128; s > 0; s >>= 1) {
    if (tid < s) red[tid] += red[tid + s];
    __syncthreads();
  }
  const float inv = 1.f / red[0];
#pragma unroll
  for (int i = 0; i < 8; ++i) P[base + tid + i * 256] = f2bf(v[i] * inv);
}

// ------------------------- GEMM 3: O = P V ---------------------------------
__global__ __launch_bounds__(256) void k_gemm_pv(
    const __bf16* __restrict__ P, const __bf16* __restrict__ vt,
    __bf16* __restrict__ ao)
{
  __shared__ __align__(16) __bf16 As[2 * ABUF];
  __shared__ __align__(16) __bf16 Bs[2 * BBUF];
  const long boffP = (long)blockIdx.z * 2048 * 2048;
  const long boffV = (long)blockIdx.z * 1024 * 2048;
  const long boffO = (long)blockIdx.z * 2048 * 1024;
  const int row0 = blockIdx.y * 128, col0 = blockIdx.x * 256;
  v8f acc[4][4] = {};
  mm_mainloop(P + boffP, 2048, vt + boffV, 2048, 2048, row0, col0, As, Bs, acc);

  const int tid = threadIdx.x, lane = tid & 31, wave = tid >> 5;
  const int wr = wave >> 2, wc = wave & 3, r16 = lane & 15, h8 = (lane >> 4) << 3;
#pragma unroll
  for (int t = 0; t < 4; ++t)
#pragma unroll
    for (int c = 0; c < 4; ++c)
#pragma unroll
      for (int j = 0; j < 8; ++j) {
        const int row = row0 + wr * 64 + t * 16 + h8 + j;
        const int col = col0 + wc * 64 + c * 16 + r16;
        ao[boffO + (long)row * 1024 + col] = f2bf(acc[t][c][j]);
      }
}

// ------------------ GEMM 4: Y = O Wp + b + x (residual) --------------------
__global__ __launch_bounds__(256) void k_gemm_proj(
    const __bf16* __restrict__ ao, const __bf16* __restrict__ wt,
    const float* __restrict__ bias, const float* __restrict__ xres,
    float* __restrict__ Y)
{
  __shared__ __align__(16) __bf16 As[2 * ABUF];
  __shared__ __align__(16) __bf16 Bs[2 * BBUF];
  const int row0 = blockIdx.y * 128, col0 = blockIdx.x * 256;
  v8f acc[4][4] = {};
  mm_mainloop(ao, 1024, wt, 1024, 1024, row0, col0, As, Bs, acc);

  const int tid = threadIdx.x, lane = tid & 31, wave = tid >> 5;
  const int wr = wave >> 2, wc = wave & 3, r16 = lane & 15, h8 = (lane >> 4) << 3;
#pragma unroll
  for (int t = 0; t < 4; ++t)
#pragma unroll
    for (int c = 0; c < 4; ++c)
#pragma unroll
      for (int j = 0; j < 8; ++j) {
        const int row = row0 + wr * 64 + t * 16 + h8 + j;
        const int col = col0 + wc * 64 + c * 16 + r16;
        const long idx = (long)row * 1024 + col;
        Y[idx] = acc[t][c][j] + bias[col] + xres[idx];
      }
}

// ------------------------- LayerNorm ---------------------------------------
__global__ __launch_bounds__(256) void k_layernorm(
    const float* __restrict__ Y, const float* __restrict__ gamma,
    const float* __restrict__ beta, float* __restrict__ out)
{
  __shared__ float red[256];
  const int tid = threadIdx.x;
  const long base = (long)blockIdx.x * 1024;
  float v[4];
  float s = 0.f;
#pragma unroll
  for (int i = 0; i < 4; ++i) { v[i] = Y[base + tid + i * 256]; s += v[i]; }
  red[tid] = s; __syncthreads();
  for (int t = 128; t > 0; t >>= 1) {
    if (tid < t) red[tid] += red[tid + t];
    __syncthreads();
  }
  const float mu = red[0] * (1.f / 1024.f); __syncthreads();
  float q = 0.f;
#pragma unroll
  for (int i = 0; i < 4; ++i) { float d = v[i] - mu; q += d * d; }
  red[tid] = q; __syncthreads();
  for (int t = 128; t > 0; t >>= 1) {
    if (tid < t) red[tid] += red[tid + t];
    __syncthreads();
  }
  const float rstd = rsqrtf(red[0] * (1.f / 1024.f) + 1e-5f);
#pragma unroll
  for (int i = 0; i < 4; ++i) {
    const int c = tid + i * 256;
    out[base + c] = (v[i] - mu) * rstd * gamma[c] + beta[c];
  }
}

// ---------------------------------------------------------------------------
extern "C" void kernel_launch(void* const* d_in, const int* in_sizes, int n_in,
                              void* d_out, int out_size, void* d_ws, size_t ws_size,
                              hipStream_t stream) {
  (void)in_sizes; (void)n_in; (void)out_size; (void)ws_size;
  const float* x      = (const float*)d_in[0];  // [4,2048,1024]
  const float* W_qkv  = (const float*)d_in[1];  // [1024,3072]
  const float* b_qkv  = (const float*)d_in[2];  // [3072]
  const float* W_proj = (const float*)d_in[3];  // [1024,1024]
  const float* b_proj = (const float*)d_in[4];  // [1024]
  const float* gamma  = (const float*)d_in[5];  // [1024]
  const float* beta   = (const float*)d_in[6];  // [1024]
  float* out = (float*)d_out;

  // workspace carve-up (~193 MB; Y aliases S which is dead after softmax)
  size_t off = 0;
  char* w = (char*)d_ws;
  auto carve = [&](size_t bytes) -> void* {
    void* p = w + off; off += (bytes + 255) & ~(size_t)255; return p;
  };
  __bf16* xb     = (__bf16*)carve(8192ull * 1024 * 2);
  __bf16* wqkvT  = (__bf16*)carve(3072ull * 1024 * 2);
  __bf16* wprojT = (__bf16*)carve(1024ull * 1024 * 2);
  __bf16* qb     = (__bf16*)carve(8192ull * 1024 * 2);
  __bf16* kb     = (__bf16*)carve(8192ull * 1024 * 2);
  __bf16* vt     = (__bf16*)carve(8192ull * 1024 * 2);
  float*  S      = (float*) carve(8192ull * 2048 * 4);
  __bf16* P      = (__bf16*)carve(8192ull * 2048 * 2);
  __bf16* ao     = (__bf16*)carve(8192ull * 1024 * 2);
  float*  Y      = S;  // alias: S is dead once P exists

  // 1) precision conversion / weight transposes
  k_cvt_bf16<<<(8192 * 1024) / 256, 256, 0, stream>>>(x, xb, 8192ll * 1024);
  k_cvt_transpose<<<(3072 * 1024) / 256, 256, 0, stream>>>(W_qkv, wqkvT, 1024, 3072);
  k_cvt_transpose<<<(1024 * 1024) / 256, 256, 0, stream>>>(W_proj, wprojT, 1024, 1024);

  // 2) QKV = x @ W_qkv + b  (M=8192, N=3072, K=1024)
  k_gemm_qkv<<<dim3(12, 64, 1), 256, 0, stream>>>(xb, wqkvT, b_qkv, qb, kb, vt);

  // 3) S = Q K^T  per batch (M=N=2048, K=1024)
  k_gemm_scores<<<dim3(8, 16, 4), 256, 0, stream>>>(qb, kb, S);

  // 4) P = softmax(S / sqrt(D))
  k_softmax<<<8192, 256, 0, stream>>>(S, P, 0.03125f /* 1/sqrt(1024) */);

  // 5) O = P V  per batch (M=2048, N=1024, K=2048)
  k_gemm_pv<<<dim3(4, 16, 4), 256, 0, stream>>>(P, vt, ao);

  // 6) Y = O @ W_proj + b_proj + x
  k_gemm_proj<<<dim3(4, 64, 1), 256, 0, stream>>>(ao, wprojT, b_proj, x, Y);

  // 7) out = LayerNorm(Y) * gamma + beta
  k_layernorm<<<8192, 256, 0, stream>>>(Y, gamma, beta, out);
}
